// MoeLayer_54013508715279
// MI455X (gfx1250) — matrix operand order repaired and checked by reading, not compiled, not run on hardware
//
#include <hip/hip_runtime.h>

#define T_TOK 32768
#define DM 1024
#define NE 8

typedef unsigned short ushort_t;
typedef __attribute__((ext_vector_type(16))) __bf16 bf16x16;
typedef __attribute__((ext_vector_type(8)))  __bf16 bf16x8;
typedef __attribute__((ext_vector_type(8)))  float  f32x8;

__device__ __forceinline__ unsigned short f2bf(float f) {
  union { float f; unsigned u; } v; v.f = f;
  unsigned r = v.u + 0x7FFFu + ((v.u >> 16) & 1u);   // round-to-nearest-even
  return (unsigned short)(r >> 16);
}
__device__ __forceinline__ float bf2f(unsigned short h) {
  union { unsigned u; float f; } v; v.u = ((unsigned)h) << 16;
  return v.f;
}

// CDNA5 async DMA: global -> LDS, 16B per lane, tracked by ASYNCcnt.
__device__ __forceinline__ void async_b128(unsigned lds_addr, unsigned gbyte_off,
                                           const ushort_t* base) {
  asm volatile("global_load_async_to_lds_b128 %0, %1, %2"
               :: "v"(lds_addr), "v"(gbyte_off), "s"(base) : "memory");
}
__device__ __forceinline__ void wait_async0() {
  asm volatile("s_wait_asynccnt 0x0" ::: "memory");
}

// ---------------- kernel 1: zero routing counters --------------------------
__global__ void moe_init_counts(int* counts) {
  if (threadIdx.x < 16) counts[threadIdx.x] = 0;
}

// ---------------- kernel 2: gate + top-2 routing ---------------------------
__global__ __launch_bounds__(256) void moe_gate_route(
    const float* __restrict__ x, const float* __restrict__ gw,
    int* __restrict__ counts, int* __restrict__ tok_list, float* __restrict__ w_list) {
  const int lane = threadIdx.x & 31;
  const int wave = threadIdx.x >> 5;
  const int tok  = blockIdx.x * 8 + wave;
  const float* xr = x + (long)tok * DM;
  float acc[NE];
#pragma unroll
  for (int e = 0; e < NE; e++) acc[e] = 0.f;
  for (int d = lane; d < DM; d += 32) {
    float xv = xr[d];
    const float* g = gw + d * NE;
#pragma unroll
    for (int e = 0; e < NE; e++) acc[e] += xv * g[e];
  }
#pragma unroll
  for (int off = 16; off > 0; off >>= 1) {
#pragma unroll
    for (int e = 0; e < NE; e++) acc[e] += __shfl_xor(acc[e], off, 32);
  }
  if (lane == 0) {
    int i0 = 0; float v0 = acc[0];
#pragma unroll
    for (int e = 1; e < NE; e++) if (acc[e] > v0) { v0 = acc[e]; i0 = e; }
    int i1 = -1; float v1 = -3.4e38f;
#pragma unroll
    for (int e = 0; e < NE; e++) if (e != i0 && acc[e] > v1) { v1 = acc[e]; i1 = e; }
    float e1  = expf(v1 - v0);          // v0 >= v1, numerically safe
    float inv = 1.f / (1.f + e1);
    float w0 = inv, w1 = e1 * inv;
    int p0 = atomicAdd(&counts[i0], 1);           // slot 0 (top-1) lists
    tok_list[i0 * T_TOK + p0] = tok;
    w_list [i0 * T_TOK + p0] = w0;
    int p1 = atomicAdd(&counts[NE + i1], 1);      // slot 1 (top-2) lists
    tok_list[(NE + i1) * T_TOK + p1] = tok;
    w_list [(NE + i1) * T_TOK + p1] = w1;
  }
}

// ---------------- kernel 3: fp32 weights -> bf16 hi/lo, transposed [e][n][k]
__global__ __launch_bounds__(256) void moe_split_w(
    const float* __restrict__ w, ushort_t* __restrict__ wh, ushort_t* __restrict__ wl) {
  __shared__ float tl[32][33];
  const int e  = blockIdx.z;
  const int kb = blockIdx.y * 32;
  const int nb = blockIdx.x * 32;
  const int tx = threadIdx.x, ty = threadIdx.y;   // 32 x 8
  const float* we = w + (long)e * DM * DM;
#pragma unroll
  for (int i = 0; i < 4; i++)
    tl[ty + 8 * i][tx] = we[(long)(kb + ty + 8 * i) * DM + nb + tx];  // coalesced over n
  __syncthreads();
  ushort_t* whe = wh + (long)e * DM * DM;
  ushort_t* wle = wl + (long)e * DM * DM;
#pragma unroll
  for (int i = 0; i < 4; i++) {
    int n = ty + 8 * i;
    float v = tl[tx][n];
    unsigned short h = f2bf(v);
    unsigned short l = f2bf(v - bf2f(h));
    long o = (long)(nb + n) * DM + kb + tx;       // coalesced over k
    whe[o] = h; wle[o] = l;
  }
}

// ---------------- kernel 4/5: grouped expert GEMM, double-buffered ---------
// block tile: 64 tokens x 256 cols; 8 waves in 2x4 grid, each wave 32x64
// -> 24 ds_load_b128 per 24 WMMA per wave per K-chunk (1.0 wmma/ds)
#define WMMA_BF16(A, B, C) \
  __builtin_amdgcn_wmma_f32_16x16x32_bf16(false, (A), false, (B), (short)0, (C), false, false)

template <int ACCUM>
__global__ __launch_bounds__(256) void moe_gemm(
    const float* __restrict__ x, const ushort_t* __restrict__ wh,
    const ushort_t* __restrict__ wl, const float* __restrict__ bias,
    const int* __restrict__ counts, const int* __restrict__ tok_list,
    const float* __restrict__ w_list, float* __restrict__ out, int slot) {
  const int e   = blockIdx.z;
  const int cnt = counts[slot * NE + e];
  const int m0  = blockIdx.y * 64;
  if (m0 >= cnt) return;                 // uniform: EXEC stays all-1s for WMMA
  const int n0 = blockIdx.x * 256;

  __shared__ int      s_tok[64];
  __shared__ float    s_w[64];
  __shared__ ushort_t Ah[2][64][40];     // 80B row stride: 16B-aligned frags
  __shared__ ushort_t Al[2][64][40];
  __shared__ ushort_t Bh[2][256][48];    // 96B row stride: 32B-aligned frags
  __shared__ ushort_t Bl[2][256][48];

  const int t    = threadIdx.x;
  const int list = slot * NE + e;
  if (t < 64) {
    int m  = m0 + t;
    int mm = m < cnt ? m : cnt - 1;      // clamp tail rows (stores are guarded)
    s_tok[t] = tok_list[list * T_TOK + mm];
    s_w[t]   = w_list [list * T_TOK + mm];
  }
  __syncthreads();

  const int wave = t >> 5, lane = t & 31;
  const int wm = wave & 1, wn = wave >> 1;        // 2x4 wave grid, 32x64 / wave
  const int half = lane >> 4, l16 = lane & 15;

  const int  arow = t >> 2,  acol = (t & 3) * 8;  // A: 64 rows x 32 cols fp32
  const long xbase = (long)s_tok[arow] * DM;
  const int  brow = t;                            // B: 256 rows(n) x 32 cols(k)
  const unsigned wbyte = (unsigned)((((long)e * DM + (n0 + brow)) * DM) * 2);

  // raw LDS byte addresses for the async DMA destinations (per stage)
  const unsigned ldsBh0 = (unsigned)(uintptr_t)&Bh[0][brow][0];
  const unsigned ldsBh1 = (unsigned)(uintptr_t)&Bh[1][brow][0];
  const unsigned ldsBl0 = (unsigned)(uintptr_t)&Bl[0][brow][0];
  const unsigned ldsBl1 = (unsigned)(uintptr_t)&Bl[1][brow][0];

  f32x8 acc[2][4] = {};
  float ar[8];

#define ISSUE_B(stage, kcn)                                                 \
  {                                                                         \
    unsigned go  = wbyte + (unsigned)((kcn) * 2);                           \
    unsigned lh  = (stage) ? ldsBh1 : ldsBh0;                               \
    unsigned ll  = (stage) ? ldsBl1 : ldsBl0;                               \
    async_b128(lh,      go,      wh);                                      \
    async_b128(lh + 16, go + 16, wh);                                      \
    async_b128(lh + 32, go + 32, wh);                                      \
    async_b128(lh + 48, go + 48, wh);                                      \
    async_b128(ll,      go,      wl);                                      \
    async_b128(ll + 16, go + 16, wl);                                      \
    async_b128(ll + 32, go + 32, wl);                                      \
    async_b128(ll + 48, go + 48, wl);                                      \
  }
#define LOAD_A(kcn)                                                         \
  _Pragma("unroll") for (int i = 0; i < 8; i++) ar[i] = x[xbase + (kcn) + acol + i];
#define STORE_A(stage)                                                      \
  _Pragma("unroll") for (int i = 0; i < 8; i++) {                           \
    ushort_t h = f2bf(ar[i]);                                               \
    Ah[stage][arow][acol + i] = h;                                          \
    Al[stage][arow][acol + i] = f2bf(ar[i] - bf2f(h));                      \
  }

  // ---- prologue: fill stage 0, prefetch A(32) into regs -------------------
  ISSUE_B(0, 0);
  LOAD_A(0);
  STORE_A(0);
  LOAD_A(32);
  wait_async0();
  __syncthreads();

  for (int kc = 0; kc < DM; kc += 32) {
    const int st = (kc >> 5) & 1;
    const bool more = (kc + 32) < DM;
    if (more) {
      ISSUE_B(st ^ 1, kc + 32);    // DMA next B tiles while we compute
      STORE_A(st ^ 1);             // A(kc+32) already in regs -> LDS
    }
    if (kc + 64 < DM) LOAD_A(kc + 64);   // global latency spans whole compute

    // ---- compute on stage st ----------------------------------------------
    // A fragments (2 M-subtiles): lane row = 32*wm + 16*mi + l16,
    // K run [16*half, +16) contiguous
    bf16x16 ahf[2], alf[2];
#pragma unroll
    for (int mi = 0; mi < 2; mi++) {
      int am = 32 * wm + 16 * mi + l16;
      bf16x8 a0 = *(const bf16x8*)&Ah[st][am][half * 16];
      bf16x8 a1 = *(const bf16x8*)&Ah[st][am][half * 16 + 8];
      ahf[mi] = __builtin_shufflevector(a0, a1, 0,1,2,3,4,5,6,7,8,9,10,11,12,13,14,15);
      bf16x8 a2 = *(const bf16x8*)&Al[st][am][half * 16];
      bf16x8 a3 = *(const bf16x8*)&Al[st][am][half * 16 + 8];
      alf[mi] = __builtin_shufflevector(a2, a3, 0,1,2,3,4,5,6,7,8,9,10,11,12,13,14,15);
    }

#pragma unroll
    for (int s = 0; s < 4; s++) {
      // B fragment: lane = col N=l16, K run [16*half, +16) contiguous
      int bn = 64 * wn + 16 * s + l16;
      bf16x16 bhf = *(const bf16x16*)&Bh[st][bn][half * 16];
      bf16x16 blf = *(const bf16x16*)&Bl[st][bn][half * 16];
#pragma unroll
      for (int mi = 0; mi < 2; mi++) {
        acc[mi][s] = WMMA_BF16(ahf[mi], bhf, acc[mi][s]);   // hi*hi
        acc[mi][s] = WMMA_BF16(ahf[mi], blf, acc[mi][s]);   // hi*lo
        acc[mi][s] = WMMA_BF16(alf[mi], bhf, acc[mi][s]);   // lo*hi
      }
    }

    if (more) wait_async0();       // next stage's DMA done (spans compute)
    __syncthreads();
  }

  // epilogue: y = gate_weight * (acc + bias); slot0 stores, slot1 accumulates
#pragma unroll
  for (int s = 0; s < 4; s++) {
    int col  = n0 + 64 * wn + 16 * s + l16;
    float bv = bias[e * DM + col];
#pragma unroll
    for (int mi = 0; mi < 2; mi++) {
#pragma unroll
      for (int r = 0; r < 8; r++) {
        int mrow = 32 * wm + 16 * mi + 8 * half + r;  // C layout: VGPR r -> row 8*half+r
        if (m0 + mrow < cnt) {
          long o  = (long)s_tok[mrow] * DM + col;
          float v = s_w[mrow] * (acc[mi][s][r] + bv);
          if (ACCUM) out[o] += v; else out[o] = v;
        }
      }
    }
  }
#undef ISSUE_B
#undef LOAD_A
#undef STORE_A
}

// ---------------------------------------------------------------------------
extern "C" void kernel_launch(void* const* d_in, const int* in_sizes, int n_in,
                              void* d_out, int out_size, void* d_ws, size_t ws_size,
                              hipStream_t stream) {
  const float* x  = (const float*)d_in[0];   // [8,4096,1024]
  const float* gw = (const float*)d_in[1];   // [1024,8]
  const float* ew = (const float*)d_in[2];   // [8,1024,1024]
  const float* eb = (const float*)d_in[3];   // [8,1024]
  float* out = (float*)d_out;

  // workspace layout (~36.3 MB total):
  char* ws = (char*)d_ws;
  int*      counts   = (int*)ws;                                   // 16 ints
  int*      tok_list = (int*)(ws + 256);                           // 16*T ints  (2MB)
  float*    w_list   = (float*)(ws + 256 + 16L * T_TOK * 4);       // 2MB
  ushort_t* wh       = (ushort_t*)(ws + 256 + 16L * T_TOK * 8);    // 16MB
  ushort_t* wl       = wh + (long)NE * DM * DM;                    // 16MB

  hipLaunchKernelGGL(moe_init_counts, dim3(1), dim3(32), 0, stream, counts);
  hipLaunchKernelGGL(moe_gate_route, dim3(T_TOK / 8), dim3(256), 0, stream,
                     x, gw, counts, tok_list, w_list);
  hipLaunchKernelGGL(moe_split_w, dim3(32, 32, 8), dim3(32, 8), 0, stream, ew, wh, wl);
  // pass A: top-1 partition of all tokens -> plain stores (fully writes d_out)
  hipLaunchKernelGGL((moe_gemm<0>), dim3(4, 512, 8), dim3(256), 0, stream,
                     x, wh, wl, eb, counts, tok_list, w_list, out, 0);
  // pass B: top-2 partition -> non-atomic accumulate (each token once per slot)
  hipLaunchKernelGGL((moe_gemm<1>), dim3(4, 512, 8), dim3(256), 0, stream,
                     x, wh, wl, eb, counts, tok_list, w_list, out, 1);
}
